// MVGRLModel_30339648979290
// MI455X (gfx1250) — matrix operand order repaired and checked by reading, not compiled, not run on hardware
//
#include <hip/hip_runtime.h>
#include <stdint.h>

// ---------------------------------------------------------------------------
// CDNA5 / gfx1250 MVGRL forward.  wave32.  WMMA bf16 16x16x32 for all GEMMs.
// ---------------------------------------------------------------------------

typedef __attribute__((ext_vector_type(16))) __bf16 v16bf;
typedef __attribute__((ext_vector_type(8)))  float  v8f;

union BF16x16 { v16bf v; unsigned short u[16]; };

__device__ __forceinline__ unsigned short f2bf(float f) {
    unsigned int u = __float_as_uint(f);
    unsigned int r = u + 0x7FFFu + ((u >> 16) & 1u);   // round-to-nearest-even
    return (unsigned short)(r >> 16);
}

// ---------------------------------------------------------------------------
// Pre-swizzle W[K x 128] (f32, row major) into per-lane WMMA bf16 B fragments.
// Fragment layout (16x16x32 bf16 B, 32x16): lane = n (0..15) in each half-wave,
// elem i<8 -> K = kt*32 + half*8 + i, elem i>=8 -> K = kt*32 + 16 + half*8 + (i-8).
// out index = ((kt*8 + nt)*32 + lane)*16 + i
// ---------------------------------------------------------------------------
__global__ void swizzle_w_kernel(const float* __restrict__ W,
                                 unsigned short* __restrict__ out, int K) {
    int idx = blockIdx.x * blockDim.x + threadIdx.x;
    int total = K * 128;
    if (idx >= total) return;
    int i    = idx & 15;
    int lane = (idx >> 4) & 31;
    int nt   = (idx >> 9) & 7;
    int kt   = idx >> 12;
    int n = nt * 16 + (lane & 15);
    int k = kt * 32 + ((lane < 16) ? 0 : 8) + (i & 7) + ((i & 8) << 1);
    out[idx] = f2bf(W[(size_t)k * 128 + n]);
}

// ---------------------------------------------------------------------------
// GEMM: C[M x 128] = op(A)[M x K] @ B, B pre-swizzled bf16.
// op = identity, or PReLU(alpha) when alpha_p != nullptr (layer-2 fusion).
// One wave per 16-row M tile, 4 waves per block.  M % 16 == 0.
// A-fragment layout (16x32 bf16): lane m = lane&15, half = lane>>4;
//   elems 0..7  -> K = k0 + half*8 + i      (contiguous 8 floats)
//   elems 8..15 -> K = k0 + 16 + half*8 + i (contiguous 8 floats)
// ---------------------------------------------------------------------------
template <int K>
__global__ __launch_bounds__(128)
void gemm_bf16_kernel(const float* __restrict__ A,
                      const unsigned short* __restrict__ Bswz,
                      float* __restrict__ C, int M,
                      const float* __restrict__ alpha_p) {
    const int lane = threadIdx.x & 31;
    const int wave = threadIdx.x >> 5;
    const int mt   = blockIdx.x * 4 + wave;
    if (mt * 16 >= M) return;                 // wave-uniform: EXEC stays all-1s

    const bool  doprelu = (alpha_p != nullptr);
    const float alpha   = doprelu ? *alpha_p : 0.0f;

    const int half = (lane >> 4) & 1;
    const int m    = lane & 15;
    const float* arow = A + (size_t)(mt * 16 + m) * K;

    v8f acc[8];
    const v8f vzero = {0.f, 0.f, 0.f, 0.f, 0.f, 0.f, 0.f, 0.f};
#pragma unroll
    for (int t = 0; t < 8; ++t) acc[t] = vzero;

    constexpr int KT = K >> 5;
#pragma unroll
    for (int kt = 0; kt < KT; ++kt) {
        const int k0 = kt * 32;
        BF16x16 a;
        const float* p0 = arow + k0 + half * 8;
        const float* p1 = arow + k0 + 16 + half * 8;
#pragma unroll
        for (int i = 0; i < 8; ++i) {
            float f = p0[i];
            f = (doprelu && f < 0.0f) ? f * alpha : f;
            a.u[i] = f2bf(f);
        }
#pragma unroll
        for (int i = 0; i < 8; ++i) {
            float f = p1[i];
            f = (doprelu && f < 0.0f) ? f * alpha : f;
            a.u[8 + i] = f2bf(f);
        }
        const unsigned short* bbase = Bswz + (size_t)kt * 8 * 512 + (size_t)lane * 16;
#pragma unroll
        for (int nt = 0; nt < 8; ++nt) {
            v16bf b = *(const v16bf*)(bbase + nt * 512);   // 32B aligned
            acc[nt] = __builtin_amdgcn_wmma_f32_16x16x32_bf16(
                false, a.v, false, b, (short)0, acc[nt], false, false);
        }
    }

    // D layout: vgpr j -> M = mt*16 + half*8 + j ; N = nt*16 + (lane&15)
    const int n0 = lane & 15;
    float* crow = C + (size_t)(mt * 16 + half * 8) * 128 + n0;
#pragma unroll
    for (int nt = 0; nt < 8; ++nt)
#pragma unroll
        for (int j = 0; j < 8; ++j)
            crow[(size_t)j * 128 + nt * 16] = acc[nt][j];
}

// ---------------------------------------------------------------------------
// Degree / normalization
// ---------------------------------------------------------------------------
__global__ void init_deg_kernel(float* __restrict__ deg, int N) {
    int i = blockIdx.x * blockDim.x + threadIdx.x;
    if (i < N) deg[i] = 1.0f;                 // self-loop contribution
}
__global__ void deg_scatter_kernel(const int* __restrict__ col,
                                   const float* __restrict__ ew,
                                   float* __restrict__ deg, int E) {
    int e = blockIdx.x * blockDim.x + threadIdx.x;
    if (e < E) atomicAdd(&deg[col[e]], ew ? ew[e] : 1.0f);
}
__global__ void deg_to_dinv_kernel(float* __restrict__ deg, int N) {
    int i = blockIdx.x * blockDim.x + threadIdx.x;
    if (i < N) deg[i] = rsqrtf(deg[i]);       // deg >= 1 always
}

// ---------------------------------------------------------------------------
// Edge message scatter: out[col] += dinv[row]*w*dinv[col] * t[perm(row)]
// One wave per edge; each lane owns one float4 (128 floats / 32 lanes).
// ---------------------------------------------------------------------------
__global__ void edge_scatter_kernel(const float* __restrict__ t,
                                    const int* __restrict__ row,
                                    const int* __restrict__ col,
                                    const float* __restrict__ ew,
                                    const float* __restrict__ dinv,
                                    const int* __restrict__ perm,
                                    float* __restrict__ out, int E) {
    int gid  = blockIdx.x * blockDim.x + threadIdx.x;
    int e    = gid >> 5;
    int lane = gid & 31;
    if (e >= E) return;
    int r = row[e], c = col[e];
    float w    = ew ? ew[e] : 1.0f;
    float coef = dinv[r] * w * dinv[c];
    int sr = perm ? perm[r] : r;
    float4 v = ((const float4*)(t + (size_t)sr * 128))[lane];
    float* dst = out + (size_t)c * 128 + lane * 4;
    atomicAdd(dst + 0, v.x * coef);
    atomicAdd(dst + 1, v.y * coef);
    atomicAdd(dst + 2, v.z * coef);
    atomicAdd(dst + 3, v.w * coef);
}

// h[v,f] += t[perm(v),f] * dinv[v]^2 + bias[f]
__global__ void finalize_kernel(float* __restrict__ h,
                                const float* __restrict__ t,
                                const int* __restrict__ perm,
                                const float* __restrict__ dinv,
                                const float* __restrict__ bias, int N) {
    int idx = blockIdx.x * blockDim.x + threadIdx.x;
    if (idx >= N * 128) return;
    int v = idx >> 7;
    int f = idx & 127;
    int s = perm ? perm[v] : v;
    float d = dinv[v];
    h[idx] += t[(size_t)s * 128 + f] * d * d + bias[f];
}

// ---------------------------------------------------------------------------
// Column sums of h1 and h2 (for the mean readout).  128 threads per block.
// ---------------------------------------------------------------------------
__global__ void col_sums_kernel(const float* __restrict__ h1,
                                const float* __restrict__ h2,
                                float* __restrict__ sums, int N) {
    int j  = threadIdx.x;                     // 0..127
    int r0 = blockIdx.x * 256;
    int r1 = r0 + 256; if (r1 > N) r1 = N;
    float s1 = 0.f, s2 = 0.f;
    for (int r = r0; r < r1; ++r) {
        s1 += h1[(size_t)r * 128 + j];
        s2 += h2[(size_t)r * 128 + j];
    }
    atomicAdd(&sums[j], s1);
    atomicAdd(&sums[128 + j], s2);
}

// r = prelu(mean(sums) @ Wro, a_ro)  (r in d_out);  v = Wbil @ r  (workspace)
__global__ void readout_kernel(const float* __restrict__ sums,
                               const float* __restrict__ Wro,
                               const float* __restrict__ aro_p,
                               const float* __restrict__ Wbil,
                               float* __restrict__ r_out,
                               float* __restrict__ v_out, int N) {
    __shared__ float rsh[128];
    int j = threadIdx.x;
    float inv_n = 1.0f / (float)N;
    float acc = 0.f;
    for (int k = 0; k < 256; ++k)
        acc += (sums[k] * inv_n) * Wro[(size_t)k * 128 + j];
    float aro = *aro_p;
    float r = (acc >= 0.f) ? acc : aro * acc;
    r_out[j] = r;
    rsh[j] = r;
    __syncthreads();
    float v = 0.f;
    for (int k = 0; k < 128; ++k)
        v += Wbil[(size_t)j * 128 + k] * rsh[k];
    v_out[j] = v;
}

// disc[node] = dot(h[node,:], v) + bbil ; one wave per node, shfl reduction
__global__ void bilinear_kernel(const float* __restrict__ h,
                                const float* __restrict__ v,
                                const float* __restrict__ bbil_p,
                                float* __restrict__ out, int N) {
    int gid  = blockIdx.x * blockDim.x + threadIdx.x;
    int node = gid >> 5;
    int lane = gid & 31;
    if (node >= N) return;
    float4 hv = ((const float4*)(h + (size_t)node * 128))[lane];
    float4 vv = ((const float4*)v)[lane];
    float s = hv.x * vv.x + hv.y * vv.y + hv.z * vv.z + hv.w * vv.w;
#pragma unroll
    for (int off = 16; off > 0; off >>= 1)
        s += __shfl_xor(s, off, 32);
    if (lane == 0) out[node] = s + *bbil_p;
}

// ---------------------------------------------------------------------------

static inline int cdiv(long long a, long long b) { return (int)((a + b - 1) / b); }

extern "C" void kernel_launch(void* const* d_in, const int* in_sizes, int n_in,
                              void* d_out, int out_size, void* d_ws, size_t ws_size,
                              hipStream_t stream) {
    const float* x    = (const float*)d_in[0];
    const int*   ei   = (const int*)d_in[1];
    const int*   dei  = (const int*)d_in[2];
    const float* dew  = (const float*)d_in[3];
    const int*   cidx = (const int*)d_in[4];
    const float* W1r  = (const float*)d_in[5];
    const float* b1r  = (const float*)d_in[6];
    const float* W2r  = (const float*)d_in[7];
    const float* b2r  = (const float*)d_in[8];
    const float* a_r  = (const float*)d_in[9];
    const float* W1d  = (const float*)d_in[10];
    const float* b1d  = (const float*)d_in[11];
    const float* W2d  = (const float*)d_in[12];
    const float* b2d  = (const float*)d_in[13];
    const float* a_d  = (const float*)d_in[14];
    const float* Wro  = (const float*)d_in[15];
    const float* a_ro = (const float*)d_in[16];
    const float* Wbil = (const float*)d_in[17];
    const float* bbil = (const float*)d_in[18];

    const int N = in_sizes[4];         // 50000
    const int E = in_sizes[3];         // 800000
    const int F = in_sizes[0] / N;     // 256 (= layer-1 K)
    (void)F;
    const size_t BIG = (size_t)N * 128;

    const int* row  = ei;        const int* col  = ei + E;
    const int* drow = dei;       const int* dcol = dei + E;

    // ---- workspace carve-up (~180 MB) ----
    float* ws   = (float*)d_ws;
    float* t1r  = ws + 0 * BIG;  // recycled later as h2_corrupt_real accumulator
    float* t1d  = ws + 1 * BIG;  // recycled later as h2_corrupt_diff accumulator
    float* h1cr = ws + 2 * BIG;
    float* h1cd = ws + 3 * BIG;
    float* t2   = ws + 4 * BIG;  // shared layer-2 GEMM scratch (sequential use)
    float* h2r  = ws + 5 * BIG;
    float* h2d  = ws + 6 * BIG;
    float* small = ws + 7 * BIG;
    float* dinvr  = small;
    float* dinvd  = small + N;
    float* sums_r = small + 2 * N;
    float* sums_d = sums_r + 256;
    float* v1     = sums_d + 256;
    float* v2     = v1 + 128;
    unsigned short* wz =
        (unsigned short*)(((uintptr_t)(v2 + 128) + 31) & ~(uintptr_t)31);
    unsigned short* w1r_z = wz;            // 256*128 elems
    unsigned short* w1d_z = wz + 32768;
    unsigned short* w2r_z = wz + 65536;    // 128*128 elems
    unsigned short* w2d_z = wz + 81920;

    // ---- output carve-up ----
    float* out   = (float*)d_out;
    float* disc  = out;                        // [4N]
    float* r1o   = out + 4 * (size_t)N;        // [128]
    float* r2o   = r1o + 128;                  // [128]
    float* h1r_o = r2o + 128;                  // [N,128]
    float* h1d_o = h1r_o + BIG;                // [N,128]

    // 1) pre-swizzle weights to bf16 WMMA fragments
    swizzle_w_kernel<<<cdiv(256 * 128, 256), 256, 0, stream>>>(W1r, w1r_z, 256);
    swizzle_w_kernel<<<cdiv(256 * 128, 256), 256, 0, stream>>>(W1d, w1d_z, 256);
    swizzle_w_kernel<<<cdiv(128 * 128, 256), 256, 0, stream>>>(W2r, w2r_z, 128);
    swizzle_w_kernel<<<cdiv(128 * 128, 256), 256, 0, stream>>>(W2d, w2d_z, 128);

    // 2) degrees -> dinv
    const int nb = cdiv(N, 256);
    init_deg_kernel<<<nb, 256, 0, stream>>>(dinvr, N);
    init_deg_kernel<<<nb, 256, 0, stream>>>(dinvd, N);
    deg_scatter_kernel<<<cdiv(E, 256), 256, 0, stream>>>(col, nullptr, dinvr, E);
    deg_scatter_kernel<<<cdiv(E, 256), 256, 0, stream>>>(dcol, dew, dinvd, E);
    deg_to_dinv_kernel<<<nb, 256, 0, stream>>>(dinvr, N);
    deg_to_dinv_kernel<<<nb, 256, 0, stream>>>(dinvd, N);

    // 3) layer-1 GEMMs (xc@W == (x@W)[perm] -> only 2 big GEMMs needed)
    const int gemm_blocks = cdiv(N / 16, 4);
    gemm_bf16_kernel<256><<<gemm_blocks, 128, 0, stream>>>(x, w1r_z, t1r, N, nullptr);
    gemm_bf16_kernel<256><<<gemm_blocks, 128, 0, stream>>>(x, w1d_z, t1d, N, nullptr);

    // 4) layer-1 scatters + finalize
    const int sb = cdiv((long long)E * 32, 256);   // edge-scatter blocks
    const int fb = cdiv((long long)N * 128, 256);  // finalize blocks
    hipMemsetAsync(h1r_o, 0, BIG * 4, stream);
    hipMemsetAsync(h1d_o, 0, BIG * 4, stream);
    hipMemsetAsync(h1cr, 0, BIG * 4, stream);
    hipMemsetAsync(h1cd, 0, BIG * 4, stream);
    edge_scatter_kernel<<<sb, 256, 0, stream>>>(t1r, row, col, nullptr, dinvr, nullptr, h1r_o, E);
    finalize_kernel<<<fb, 256, 0, stream>>>(h1r_o, t1r, nullptr, dinvr, b1r, N);
    edge_scatter_kernel<<<sb, 256, 0, stream>>>(t1d, drow, dcol, dew, dinvd, nullptr, h1d_o, E);
    finalize_kernel<<<fb, 256, 0, stream>>>(h1d_o, t1d, nullptr, dinvd, b1d, N);
    edge_scatter_kernel<<<sb, 256, 0, stream>>>(t1r, row, col, nullptr, dinvr, cidx, h1cr, E);
    finalize_kernel<<<fb, 256, 0, stream>>>(h1cr, t1r, cidx, dinvr, b1r, N);
    edge_scatter_kernel<<<sb, 256, 0, stream>>>(t1d, drow, dcol, dew, dinvd, cidx, h1cd, E);
    finalize_kernel<<<fb, 256, 0, stream>>>(h1cd, t1d, cidx, dinvd, b1d, N);

    // 5) layer-2: GEMM (PReLU fused) -> scatter -> finalize, for 4 paths
    // real
    gemm_bf16_kernel<128><<<gemm_blocks, 128, 0, stream>>>(h1r_o, w2r_z, t2, N, a_r);
    hipMemsetAsync(h2r, 0, BIG * 4, stream);
    edge_scatter_kernel<<<sb, 256, 0, stream>>>(t2, row, col, nullptr, dinvr, nullptr, h2r, E);
    finalize_kernel<<<fb, 256, 0, stream>>>(h2r, t2, nullptr, dinvr, b2r, N);
    // diff
    gemm_bf16_kernel<128><<<gemm_blocks, 128, 0, stream>>>(h1d_o, w2d_z, t2, N, a_d);
    hipMemsetAsync(h2d, 0, BIG * 4, stream);
    edge_scatter_kernel<<<sb, 256, 0, stream>>>(t2, drow, dcol, dew, dinvd, nullptr, h2d, E);
    finalize_kernel<<<fb, 256, 0, stream>>>(h2d, t2, nullptr, dinvd, b2d, N);
    // corrupted real  (t1r buffer recycled as the h2 accumulator)
    float* h2cr = t1r;
    gemm_bf16_kernel<128><<<gemm_blocks, 128, 0, stream>>>(h1cr, w2r_z, t2, N, a_r);
    hipMemsetAsync(h2cr, 0, BIG * 4, stream);
    edge_scatter_kernel<<<sb, 256, 0, stream>>>(t2, row, col, nullptr, dinvr, nullptr, h2cr, E);
    finalize_kernel<<<fb, 256, 0, stream>>>(h2cr, t2, nullptr, dinvr, b2r, N);
    // corrupted diff (t1d buffer recycled)
    float* h2cd = t1d;
    gemm_bf16_kernel<128><<<gemm_blocks, 128, 0, stream>>>(h1cd, w2d_z, t2, N, a_d);
    hipMemsetAsync(h2cd, 0, BIG * 4, stream);
    edge_scatter_kernel<<<sb, 256, 0, stream>>>(t2, drow, dcol, dew, dinvd, nullptr, h2cd, E);
    finalize_kernel<<<fb, 256, 0, stream>>>(h2cd, t2, nullptr, dinvd, b2d, N);

    // 6) readouts r1, r2 and bilinear projection vectors v1, v2
    hipMemsetAsync(sums_r, 0, 512 * 4, stream);  // sums_r[256] + sums_d[256]
    col_sums_kernel<<<cdiv(N, 256), 128, 0, stream>>>(h1r_o, h2r, sums_r, N);
    col_sums_kernel<<<cdiv(N, 256), 128, 0, stream>>>(h1d_o, h2d, sums_d, N);
    readout_kernel<<<1, 128, 0, stream>>>(sums_r, Wro, a_ro, Wbil, r1o, v1, N);
    readout_kernel<<<1, 128, 0, stream>>>(sums_d, Wro, a_ro, Wbil, r2o, v2, N);

    // 7) discriminator: [bil(h2d,r1), bil(h2r,r2), bil(h4d,r1), bil(h4r,r2)]
    const int bb = cdiv((long long)N * 32, 256);
    bilinear_kernel<<<bb, 256, 0, stream>>>(h2d,  v1, bbil, disc + 0 * (size_t)N, N);
    bilinear_kernel<<<bb, 256, 0, stream>>>(h2r,  v2, bbil, disc + 1 * (size_t)N, N);
    bilinear_kernel<<<bb, 256, 0, stream>>>(h2cd, v1, bbil, disc + 2 * (size_t)N, N);
    bilinear_kernel<<<bb, 256, 0, stream>>>(h2cr, v2, bbil, disc + 3 * (size_t)N, N);
}
